// SSDHead_71373766525588
// MI455X (gfx1250) — matrix operand order repaired and checked by reading, not compiled, not run on hardware
//
#include <hip/hip_runtime.h>
#include <hip/hip_bf16.h>

typedef __attribute__((ext_vector_type(16))) _Float16 v16h;
typedef __attribute__((ext_vector_type(8)))  _Float16 v8h;
typedef __attribute__((ext_vector_type(8)))  float    v8f;

#define NUM_CLASSES 21
#define NTILES 10            // 160 output channels (24 loc + 126 conf + 10 pad)
#define NCH 160
#define NCH_PAD 161          // epilogue LDS row stride (conflict-free)
#define BROW_H 40            // staged B row stride in halfs (80 B, conflict-free)
#define BBUF_H 6400          // one staged B buffer: 160 rows * 40 halfs

// ---------------------------------------------------------------------------
// Pack 1: NCHW f32 -> NHWC f16, tiled transpose through LDS.
// ---------------------------------------------------------------------------
__global__ __launch_bounds__(256)
void pack_feat_kernel(const float* __restrict__ in,
                      _Float16* __restrict__ out,
                      int C, int HW, int tilesHW) {
    __shared__ float tile[32][33];
    const int b   = blockIdx.y;
    const int thw = blockIdx.x % tilesHW;
    const int tc  = blockIdx.x / tilesHW;
    const int lx  = threadIdx.x & 31;
    const int ly  = threadIdx.x >> 5;

#pragma unroll
    for (int r = 0; r < 4; ++r) {
        const int c  = tc * 32 + ly + r * 8;
        const int hw = thw * 32 + lx;
        float v = 0.0f;
        if (c < C && hw < HW) v = in[((size_t)b * C + c) * HW + hw];
        tile[ly + r * 8][lx] = v;
    }
    __syncthreads();
#pragma unroll
    for (int r = 0; r < 4; ++r) {
        const int hw = thw * 32 + ly + r * 8;
        const int c  = tc * 32 + lx;
        if (c < C && hw < HW)
            out[((size_t)b * HW + hw) * C + c] = (_Float16)tile[lx][ly + r * 8];
    }
}

// ---------------------------------------------------------------------------
// Pack 2: weights OIHW f32 -> combined [160][9*C] f16, K ordered (tap, ci)
// ---------------------------------------------------------------------------
__global__ void pack_w_kernel(const float* __restrict__ locw,
                              const float* __restrict__ confw,
                              _Float16* __restrict__ wr,
                              int C, int n) {
    int i = blockIdx.x * blockDim.x + threadIdx.x;
    if (i >= n) return;
    int K  = 9 * C;
    int k  = i % K;
    int nn = i / K;
    int t  = k / C;
    int ci = k % C;
    float v = 0.0f;
    if (nn < 24)       v = locw [((size_t)nn        * C + ci) * 9 + t];
    else if (nn < 150) v = confw[((size_t)(nn - 24) * C + ci) * 9 + t];
    wr[i] = (_Float16)v;
}

// ---------------------------------------------------------------------------
// Implicit-GEMM conv (both heads fused, N=160) + softmax + box decode.
// Block = 128 threads = 4 waves; each wave owns TWO 16-row M-tiles.
// B tile (160x32 halfs) staged cooperatively: global->regs (prefetched one
// k-step ahead) ->ds_store-> double-buffered LDS ->ds_read_b128 fragments.
// One barrier per k-step.
// ---------------------------------------------------------------------------
__global__ __launch_bounds__(128)
void ssd_gemm_kernel(const _Float16* __restrict__ A,   // NHWC f16 feat
                     const _Float16* __restrict__ Wr,  // [160][9*C] f16
                     const float* __restrict__ locb,
                     const float* __restrict__ confb,
                     const float* __restrict__ priors,
                     int pbase,
                     float* __restrict__ out,          // [32][11640][25]
                     int H, int W, int Cin, int Mtotal) {
    const int tid  = threadIdx.x;
    const int lane = tid & 31;
    const int wave = tid >> 5;
    const bool hi  = lane >= 16;
    const int  ln  = lane & 15;
    const int  HW  = H * W;
    const int  K   = 9 * Cin;
    const int  m0  = blockIdx.x * 128 + wave * 32;

    // LDS overlay: staged B (2 x 12.8KB) during k-loop, epilogue tile after.
    __shared__ __align__(16) unsigned char lds[128 * NCH_PAD * 4];
    _Float16* const bst = (_Float16*)lds;

    // Rows this lane feeds into the two A fragments
    const int  mA = m0 + ln;
    const int  mB = m0 + 16 + ln;
    const bool vA = mA < Mtotal;
    const bool vB = mB < Mtotal;
    int bA = 0, yA = 0, xA = 0, bB = 0, yB = 0, xB = 0;
    if (vA) { bA = mA / HW; int c = mA % HW; yA = c / W; xA = c % W; }
    if (vB) { bB = mB / HW; int c = mB % HW; yB = c / W; xB = c % W; }

    v8f acc0[NTILES], acc1[NTILES];
#pragma unroll
    for (int nt = 0; nt < NTILES; ++nt) {
        acc0[nt] = (v8f){0,0,0,0,0,0,0,0};
        acc1[nt] = (v8f){0,0,0,0,0,0,0,0};
    }

    const int aoff = hi ? 8 : 0;          // A K-half select
    // Cooperative B staging assignment: thread -> (row0 = tid/4, 16B part)
    const int row0 = tid >> 2;            // 0..31 (rows row0 + 32*i, i=0..4)
    const int part = tid & 3;             // 16-byte chunk within 64B row
    const _Float16* const gst = Wr + (size_t)row0 * K + part * 8;
    const int lst = row0 * BROW_H + part * 8;   // LDS halfs offset

    const int nci = Cin >> 5;             // 32-wide K chunks per tap
    const int KS  = 9 * nci;

    // ---- prologue: prefetch k-step 0 staging chunks into registers
    v8h sr[5];
#pragma unroll
    for (int i = 0; i < 5; ++i)
        sr[i] = *(const v8h*)(gst + (size_t)(32 * i) * K);

    int t = 0, ci = 0;
    const _Float16* arowA = A;
    const _Float16* arowB = A;
    bool pA = false, pB = false;

    for (int ks = 0; ks < KS; ++ks) {
        // ---- commit staged registers into LDS buffer (ks & 1)
        _Float16* const bb = bst + (ks & 1) * BBUF_H;
#pragma unroll
        for (int i = 0; i < 5; ++i)
            *(v8h*)(bb + lst + i * 32 * BROW_H) = sr[i];
        __syncthreads();

        if (ci == 0) {  // entering a new conv tap
            const int dy = t / 3 - 1;
            const int dx = t % 3 - 1;
            const int yyA = yA + dy, xxA = xA + dx;
            const int yyB = yB + dy, xxB = xB + dx;
            pA = vA && (yyA >= 0) && (yyA < H) && (xxA >= 0) && (xxA < W);
            pB = vB && (yyB >= 0) && (yyB < H) && (xxB >= 0) && (xxB < W);
            arowA = A + ((size_t)(bA * H + yyA) * W + xxA) * Cin;
            arowB = A + ((size_t)(bB * H + yyB) * W + xxB) * Cin;
        }
        const int ci0 = ci << 5;

        // ---- prefetch next k-step's B chunks (in flight during compute)
        if (ks + 1 < KS) {
            int tn = t, cn = ci + 1;
            if (cn == nci) { cn = 0; ++tn; }
            const int kbn = tn * Cin + (cn << 5);
#pragma unroll
            for (int i = 0; i < 5; ++i)
                sr[i] = *(const v8h*)(gst + (size_t)(32 * i) * K + kbn);
        }

        // ---- A fragments (predicated per-lane; reconverges before WMMA)
        v16h af0 = (v16h){};
        v16h af1 = (v16h){};
        if (pA) {
            v8h a0 = *(const v8h*)(arowA + ci0 + aoff);
            v8h a1 = *(const v8h*)(arowA + ci0 + 16 + aoff);
            af0 = __builtin_shufflevector(a0, a1,
                    0,1,2,3,4,5,6,7,8,9,10,11,12,13,14,15);
        }
        if (pB) {
            v8h a0 = *(const v8h*)(arowB + ci0 + aoff);
            v8h a1 = *(const v8h*)(arowB + ci0 + 16 + aoff);
            af1 = __builtin_shufflevector(a0, a1,
                    0,1,2,3,4,5,6,7,8,9,10,11,12,13,14,15);
        }

        // ---- compute: fragments from LDS (conflict-free ds_read_b128 pairs)
        const _Float16* const rb = bb + ln * BROW_H + (hi ? 8 : 0);
#pragma unroll
        for (int nt = 0; nt < NTILES; ++nt) {
            v8h f0 = *(const v8h*)(rb + nt * 16 * BROW_H);
            v8h f1 = *(const v8h*)(rb + nt * 16 * BROW_H + 16);
            v16h bf = __builtin_shufflevector(f0, f1,
                    0,1,2,3,4,5,6,7,8,9,10,11,12,13,14,15);
            acc0[nt] = __builtin_amdgcn_wmma_f32_16x16x32_f16(
                false, af0, false, bf, (short)0, acc0[nt], false, false);
            acc1[nt] = __builtin_amdgcn_wmma_f32_16x16x32_f16(
                false, af1, false, bf, (short)0, acc1[nt], false, false);
        }

        if (++ci == nci) { ci = 0; ++t; }
    }

    // ---- epilogue staging reuses the same LDS region
    __syncthreads();
    float* const smem = (float*)lds;
#pragma unroll
    for (int nt = 0; nt < NTILES; ++nt) {
#pragma unroll
        for (int r = 0; r < 8; ++r) {
            const int mr = r + (hi ? 8 : 0);   // C/D layout: VGPR r -> M=r / r+8
            smem[(wave * 32      + mr) * NCH_PAD + nt * 16 + ln] = acc0[nt][r];
            smem[(wave * 32 + 16 + mr) * NCH_PAD + nt * 16 + ln] = acc1[nt][r];
        }
    }
    __syncthreads();

    // ---- 1 thread per position, 6 anchors each
    const int pos = threadIdx.x;
    const int mg  = blockIdx.x * 128 + pos;
    if (mg < Mtotal) {
        const int bb2  = mg / HW;
        const int cell = mg % HW;
        const float* row = &smem[pos * NCH_PAD];
#pragma unroll
        for (int a = 0; a < 6; ++a) {
            float l0 = row[4 * a + 0] + locb[4 * a + 0];
            float l1 = row[4 * a + 1] + locb[4 * a + 1];
            float l2 = row[4 * a + 2] + locb[4 * a + 2];
            float l3 = row[4 * a + 3] + locb[4 * a + 3];
            float e[NUM_CLASSES];
            float mx = -3.0e38f;
            for (int j = 0; j < NUM_CLASSES; ++j) {
                float v = row[24 + NUM_CLASSES * a + j] + confb[NUM_CLASSES * a + j];
                e[j] = v;
                mx = fmaxf(mx, v);
            }
            float s = 0.0f;
            for (int j = 0; j < NUM_CLASSES; ++j) { e[j] = __expf(e[j] - mx); s += e[j]; }
            const float inv = 1.0f / s;
            const int p = pbase + cell * 6 + a;
            const float* pr = priors + 4 * p;
            const float cx = pr[0], cy = pr[1], pw = pr[2], ph = pr[3];
            const float bx = cx + l0 * 0.1f * pw;
            const float by = cy + l1 * 0.1f * ph;
            const float bw = pw * __expf(l2 * 0.2f);
            const float bh = ph * __expf(l3 * 0.2f);
            float* o = out + ((size_t)bb2 * 11640 + p) * 25;
            const float mnx = bx - bw * 0.5f;
            const float mny = by - bh * 0.5f;
            o[0] = mnx; o[1] = mny; o[2] = mnx + bw; o[3] = mny + bh;
            for (int j = 0; j < NUM_CLASSES; ++j) o[4 + j] = e[j] * inv;
        }
    }
}

// ---------------------------------------------------------------------------
// Host launcher
// ---------------------------------------------------------------------------
static inline size_t align256(size_t v) { return (v + 255) & ~(size_t)255; }

extern "C" void kernel_launch(void* const* d_in, const int* in_sizes, int n_in,
                              void* d_out, int out_size, void* d_ws, size_t ws_size,
                              hipStream_t stream) {
    static const int CHs[6] = {512, 1024, 512, 256, 256, 256};
    static const int Fs[6]  = {38, 19, 10, 5, 3, 1};
    static const int PB[6]  = {0, 8664, 10830, 11430, 11580, 11634};

    char* ws = (char*)d_ws;
    size_t off = 0;
    _Float16* featH[6];
    _Float16* wr[6];
    for (int s = 0; s < 6; ++s) {
        featH[s] = (_Float16*)(ws + off);
        off += align256((size_t)32 * Fs[s] * Fs[s] * CHs[s] * sizeof(_Float16));
    }
    for (int s = 0; s < 6; ++s) {
        wr[s] = (_Float16*)(ws + off);
        off += align256((size_t)NCH * 9 * CHs[s] * sizeof(_Float16));
    }

    const float* priors = (const float*)d_in[30];
    float* out = (float*)d_out;

    for (int s = 0; s < 6; ++s) {
        const float* feat  = (const float*)d_in[5 * s + 0];
        const float* locw  = (const float*)d_in[5 * s + 1];
        const float* confw = (const float*)d_in[5 * s + 3];
        const int C = CHs[s], F = Fs[s], HW = F * F;
        const int tilesHW = (HW + 31) / 32;
        const int tilesC  = (C + 31) / 32;
        dim3 grid(tilesHW * tilesC, 32);
        pack_feat_kernel<<<grid, 256, 0, stream>>>(feat, featH[s], C, HW, tilesHW);
        const int nw = NCH * 9 * C;
        pack_w_kernel<<<(nw + 255) / 256, 256, 0, stream>>>(locw, confw, wr[s], C, nw);
    }

    for (int s = 0; s < 6; ++s) {
        const float* locb  = (const float*)d_in[5 * s + 2];
        const float* confb = (const float*)d_in[5 * s + 4];
        const int C = CHs[s], F = Fs[s], HW = F * F;
        const int M = 32 * HW;
        const int grid = (M + 127) / 128;
        ssd_gemm_kernel<<<grid, 128, 0, stream>>>(
            featH[s], wr[s], locb, confb, priors, PB[s], out, F, F, C, M);
    }
}
